// TopKMoE_54503134986828
// MI455X (gfx1250) — compile-verified, matrix operand order checked by reading, not run on hardware
//
#include <hip/hip_runtime.h>

#define N_TOK  8192
#define DIN    1024
#define DHID   4096
#define DOUT   1024
#define NEXP   8
#define TOPK   2
#define NROUTE (N_TOK * TOPK)   // 16384 routed rows total (k=2 distinct experts/token)
#define HSLICE 2048
#define NSLICES (DHID / HSLICE)

#define BM 128
#define BN 128
#define BK 32
#define LDT 48   // padded LDS stride (elements); 96B rows keep 16B alignment
#define ABUF_BYTES (BM * LDT * 2)

typedef __bf16 bf16;
typedef bf16  bf16x16 __attribute__((ext_vector_type(16)));
typedef float f32x8   __attribute__((ext_vector_type(8)));

struct alignas(16) Q4 { unsigned int x, y, z, w; };
struct alignas(16) F4 { float x, y, z, w; };
union Frag { bf16x16 v; Q4 q[2]; };

// CDNA5 async global->LDS copy (ASYNCcnt-tracked). VDST VGPR supplies the
// per-lane LDS byte address, VADDR the 64-bit global address (GV mode).
__device__ __forceinline__ void async_copy_b128(unsigned lds_byte, const void* gptr) {
  asm volatile("global_load_async_to_lds_b128 %0, %1, off"
               :: "v"(lds_byte), "v"(gptr) : "memory");
}
__device__ __forceinline__ void wait_async0() {
#if __has_builtin(__builtin_amdgcn_s_wait_asynccnt)
  __builtin_amdgcn_s_wait_asynccnt(0);
#else
  asm volatile("s_wait_asynccnt 0x0" ::: "memory");
#endif
}

// ---------------------------------------------------------------- gating ----
__global__ __launch_bounds__(256) void k_gate(const float* __restrict__ x,
    const float* __restrict__ Wg, const float* __restrict__ bg,
    int* __restrict__ top_e, float* __restrict__ top_w, int* __restrict__ counts)
{
  const int lane = threadIdx.x & 31;
  const int t = blockIdx.x * 8 + (threadIdx.x >> 5);
  const float* xr = x + (size_t)t * DIN;
  float s[NEXP];
#pragma unroll
  for (int e = 0; e < NEXP; ++e) s[e] = 0.f;
  for (int i = lane; i < DIN; i += 32) {
    float xv = xr[i];
    const float* wr = Wg + (size_t)i * NEXP;
#pragma unroll
    for (int e = 0; e < NEXP; ++e) s[e] = fmaf(xv, wr[e], s[e]);
  }
#pragma unroll
  for (int e = 0; e < NEXP; ++e) {
#pragma unroll
    for (int off = 16; off > 0; off >>= 1) s[e] += __shfl_xor(s[e], off, 32);
  }
  if (lane == 0) {
#pragma unroll
    for (int e = 0; e < NEXP; ++e) s[e] += bg[e];
    float mx = s[0];
#pragma unroll
    for (int e = 1; e < NEXP; ++e) mx = fmaxf(mx, s[e]);
    float p[NEXP], sum = 0.f;
#pragma unroll
    for (int e = 0; e < NEXP; ++e) { p[e] = __expf(s[e] - mx); sum += p[e]; }
    float inv = 1.f / sum;
#pragma unroll
    for (int e = 0; e < NEXP; ++e) p[e] *= inv;
    int i0 = 0;
#pragma unroll
    for (int e = 1; e < NEXP; ++e) if (s[e] > s[i0]) i0 = e;
    int i1 = (i0 == 0) ? 1 : 0;
#pragma unroll
    for (int e = 0; e < NEXP; ++e) if (e != i0 && s[e] > s[i1]) i1 = e;
    float den = p[i0] + p[i1] + 1e-8f;
    top_e[t * 2 + 0] = i0;          top_e[t * 2 + 1] = i1;
    top_w[t * 2 + 0] = p[i0] / den; top_w[t * 2 + 1] = p[i1] / den;
    atomicAdd(&counts[i0], 1);
    atomicAdd(&counts[i1], 1);
  }
}

__global__ void k_zero(int* counts, int* cursors) {
  if (threadIdx.x < NEXP) { counts[threadIdx.x] = 0; cursors[threadIdx.x] = 0; }
}

__global__ void k_scan(const int* __restrict__ counts, int* __restrict__ offsets) {
  if (threadIdx.x == 0) {
    int acc = 0;
    for (int e = 0; e < NEXP; ++e) { offsets[e] = acc; acc += counts[e]; }
    offsets[NEXP] = acc;
  }
}

__global__ __launch_bounds__(256) void k_assign(const int* __restrict__ top_e,
    const float* __restrict__ top_w, const int* __restrict__ offsets,
    int* __restrict__ cursors, int* __restrict__ tok_idx,
    int* __restrict__ row_of, float* __restrict__ w_of)
{
  int t = blockIdx.x * 256 + threadIdx.x;
  if (t >= N_TOK) return;
#pragma unroll
  for (int j = 0; j < TOPK; ++j) {
    int e = top_e[t * 2 + j];
    int pos = atomicAdd(&cursors[e], 1);
    int row = offsets[e] + pos;
    tok_idx[row] = t;
    row_of[t * 2 + j] = row;
    w_of[t * 2 + j] = top_w[t * 2 + j];
  }
}

__global__ __launch_bounds__(256) void k_cvt(const float* __restrict__ x,
                                             bf16* __restrict__ xbf)
{
  size_t i = ((size_t)blockIdx.x * 256 + threadIdx.x) * 4;
  F4 f = *(const F4*)(x + i);
  union { bf16 h[4]; unsigned long long u; } o;
  o.h[0] = (bf16)f.x; o.h[1] = (bf16)f.y; o.h[2] = (bf16)f.z; o.h[3] = (bf16)f.w;
  *(unsigned long long*)(xbf + i) = o.u;
}

// -------------------------------------------------- FFN1: relu(Xg@W1+b1) ----
__global__ __launch_bounds__(256) void k_ffn1(
    const bf16* __restrict__ xbf, const float* __restrict__ W1,
    const float* __restrict__ b1, const int* __restrict__ tok_idx,
    const int* __restrict__ counts, const int* __restrict__ offsets,
    bf16* __restrict__ H, int nSliceBase)
{
  const int e = blockIdx.z;
  const int cnt = counts[e];
  const int tile = blockIdx.x;
  if (tile * BM >= cnt) return;
  const int rowStart = offsets[e] + tile * BM;
  const int nBase = blockIdx.y * BN;
  const float* Wp = W1 + (size_t)e * DIN * DHID + nSliceBase + nBase;

  __shared__ bf16 As[2][BM * LDT];
  __shared__ bf16 Bs[2][BN * LDT];

  const int tid = threadIdx.x;
  const int lane = tid & 31, wave = tid >> 5;
  const int wm = (wave & 1) * 64;
  const int wn = (wave >> 1) * 32;
  const int lo = lane & 15, hi = lane >> 4;

  // A tile: 2 async 16B chunks per thread
  const bf16* aptr[2]; unsigned aLds[2]; bool av[2];
#pragma unroll
  for (int i = 0; i < 2; ++i) {
    int idx = tid + i * 256;
    int r = idx >> 2, c = (idx & 3) * 8;
    av[i] = (tile * BM + r) < cnt;
    int g = av[i] ? tok_idx[rowStart + r] : 0;
    aptr[i] = xbf + (size_t)g * DIN + c;
    aLds[i] = (unsigned)(size_t)(&As[0][r * LDT + c]);
  }
  // B tile: 4k x 4n micro-tile per thread (pack 4 k-contig bf16 -> b64 store)
  const int kg  = (tid >> 5) * 4;
  const int bn0 = (tid & 31) * 4;

  float breg[4][4];
  auto loadB = [&](int kt) {
    const float* wp = Wp + (size_t)(kt * BK + kg) * DHID + bn0;
#pragma unroll
    for (int r = 0; r < 4; ++r) {
      F4 f = *(const F4*)(wp + (size_t)r * DHID);
      breg[r][0] = f.x; breg[r][1] = f.y; breg[r][2] = f.z; breg[r][3] = f.w;
    }
  };
  auto asyncA = [&](int kt, int buf) {
    const unsigned bo = (unsigned)buf * ABUF_BYTES;
#pragma unroll
    for (int i = 0; i < 2; ++i)
      if (av[i]) async_copy_b128(aLds[i] + bo, aptr[i] + kt * BK);
  };

  const f32x8 zeroV = {0.f, 0.f, 0.f, 0.f, 0.f, 0.f, 0.f, 0.f};
  f32x8 acc[4][2];
#pragma unroll
  for (int im = 0; im < 4; ++im)
#pragma unroll
    for (int in = 0; in < 2; ++in) acc[im][in] = zeroV;

  asyncA(0, 0);
  loadB(0);
  int p = 0;
  const int KT = DIN / BK;
  for (int kt = 0; kt < KT; ++kt) {
#pragma unroll
    for (int c = 0; c < 4; ++c) {
      union { bf16 h[4]; unsigned long long u; } pk;
      pk.h[0] = (bf16)breg[0][c]; pk.h[1] = (bf16)breg[1][c];
      pk.h[2] = (bf16)breg[2][c]; pk.h[3] = (bf16)breg[3][c];
      *(unsigned long long*)(&Bs[p][(bn0 + c) * LDT + kg]) = pk.u;
    }
    wait_async0();
    __syncthreads();
    if (kt + 1 < KT) { asyncA(kt + 1, p ^ 1); loadB(kt + 1); }
    Frag a[4], b[2];
#pragma unroll
    for (int im = 0; im < 4; ++im) {
      int m = wm + im * 16 + lo;
      a[im].q[0] = *(const Q4*)(&As[p][m * LDT + 8 * hi]);
      a[im].q[1] = *(const Q4*)(&As[p][m * LDT + 16 + 8 * hi]);
    }
#pragma unroll
    for (int in = 0; in < 2; ++in) {
      int n = wn + in * 16 + lo;
      b[in].q[0] = *(const Q4*)(&Bs[p][n * LDT + 8 * hi]);
      b[in].q[1] = *(const Q4*)(&Bs[p][n * LDT + 16 + 8 * hi]);
    }
#pragma unroll
    for (int im = 0; im < 4; ++im)
#pragma unroll
      for (int in = 0; in < 2; ++in)
        acc[im][in] = __builtin_amdgcn_wmma_f32_16x16x32_bf16(
            false, a[im].v, false, b[in].v, (short)0, acc[im][in], false, false);
    p ^= 1;
  }
#pragma unroll
  for (int in = 0; in < 2; ++in) {
    int nl = wn + in * 16 + lo;
    float bias = b1[(size_t)e * DHID + nSliceBase + nBase + nl];
#pragma unroll
    for (int im = 0; im < 4; ++im) {
#pragma unroll
      for (int j = 0; j < 8; ++j) {
        int m = wm + im * 16 + hi * 8 + j;
        int r = tile * BM + m;
        if (r < cnt) {
          float v = acc[im][in][j] + bias;
          v = v > 0.f ? v : 0.f;
          H[(size_t)(rowStart + m) * HSLICE + nBase + nl] = (bf16)v;
        }
      }
    }
  }
}

// ---------------------------------------------------- FFN2: Y (+)= H@W2 ----
__global__ __launch_bounds__(256) void k_ffn2(
    const bf16* __restrict__ H, const float* __restrict__ W2,
    const float* __restrict__ b2, const int* __restrict__ counts,
    const int* __restrict__ offsets, float* __restrict__ Y,
    int kSliceBase, int first)
{
  const int e = blockIdx.z;
  const int cnt = counts[e];
  const int tile = blockIdx.x;
  if (tile * BM >= cnt) return;
  const int rowStart = offsets[e] + tile * BM;
  const int nBase = blockIdx.y * BN;
  const float* Wp = W2 + (size_t)e * DHID * DOUT + nBase;

  __shared__ bf16 As[2][BM * LDT];
  __shared__ bf16 Bs[2][BN * LDT];

  const int tid = threadIdx.x;
  const int lane = tid & 31, wave = tid >> 5;
  const int wm = (wave & 1) * 64;
  const int wn = (wave >> 1) * 32;
  const int lo = lane & 15, hi = lane >> 4;

  const bf16* aptr[2]; unsigned aLds[2]; bool av[2];
#pragma unroll
  for (int i = 0; i < 2; ++i) {
    int idx = tid + i * 256;
    int r = idx >> 2, c = (idx & 3) * 8;
    av[i] = (tile * BM + r) < cnt;
    aptr[i] = H + (size_t)(rowStart + r) * HSLICE + c;
    aLds[i] = (unsigned)(size_t)(&As[0][r * LDT + c]);
  }
  const int kg  = (tid >> 5) * 4;
  const int bn0 = (tid & 31) * 4;

  float breg[4][4];
  auto loadB = [&](int kt) {
    const float* wp = Wp + (size_t)(kSliceBase + kt * BK + kg) * DOUT + bn0;
#pragma unroll
    for (int r = 0; r < 4; ++r) {
      F4 f = *(const F4*)(wp + (size_t)r * DOUT);
      breg[r][0] = f.x; breg[r][1] = f.y; breg[r][2] = f.z; breg[r][3] = f.w;
    }
  };
  auto asyncA = [&](int kt, int buf) {
    const unsigned bo = (unsigned)buf * ABUF_BYTES;
#pragma unroll
    for (int i = 0; i < 2; ++i)
      if (av[i]) async_copy_b128(aLds[i] + bo, aptr[i] + kt * BK);
  };

  const f32x8 zeroV = {0.f, 0.f, 0.f, 0.f, 0.f, 0.f, 0.f, 0.f};
  f32x8 acc[4][2];
#pragma unroll
  for (int im = 0; im < 4; ++im)
#pragma unroll
    for (int in = 0; in < 2; ++in) acc[im][in] = zeroV;

  asyncA(0, 0);
  loadB(0);
  int p = 0;
  const int KT = HSLICE / BK;
  for (int kt = 0; kt < KT; ++kt) {
#pragma unroll
    for (int c = 0; c < 4; ++c) {
      union { bf16 h[4]; unsigned long long u; } pk;
      pk.h[0] = (bf16)breg[0][c]; pk.h[1] = (bf16)breg[1][c];
      pk.h[2] = (bf16)breg[2][c]; pk.h[3] = (bf16)breg[3][c];
      *(unsigned long long*)(&Bs[p][(bn0 + c) * LDT + kg]) = pk.u;
    }
    wait_async0();
    __syncthreads();
    if (kt + 1 < KT) { asyncA(kt + 1, p ^ 1); loadB(kt + 1); }
    Frag a[4], b[2];
#pragma unroll
    for (int im = 0; im < 4; ++im) {
      int m = wm + im * 16 + lo;
      a[im].q[0] = *(const Q4*)(&As[p][m * LDT + 8 * hi]);
      a[im].q[1] = *(const Q4*)(&As[p][m * LDT + 16 + 8 * hi]);
    }
#pragma unroll
    for (int in = 0; in < 2; ++in) {
      int n = wn + in * 16 + lo;
      b[in].q[0] = *(const Q4*)(&Bs[p][n * LDT + 8 * hi]);
      b[in].q[1] = *(const Q4*)(&Bs[p][n * LDT + 16 + 8 * hi]);
    }
#pragma unroll
    for (int im = 0; im < 4; ++im)
#pragma unroll
      for (int in = 0; in < 2; ++in)
        acc[im][in] = __builtin_amdgcn_wmma_f32_16x16x32_bf16(
            false, a[im].v, false, b[in].v, (short)0, acc[im][in], false, false);
    p ^= 1;
  }
#pragma unroll
  for (int in = 0; in < 2; ++in) {
    int nl = wn + in * 16 + lo;
    float bias = b2[(size_t)e * DOUT + nBase + nl];
#pragma unroll
    for (int im = 0; im < 4; ++im) {
#pragma unroll
      for (int j = 0; j < 8; ++j) {
        int m = wm + im * 16 + hi * 8 + j;
        int r = tile * BM + m;
        if (r < cnt) {
          float* yp = Y + (size_t)(rowStart + m) * DOUT + nBase + nl;
          if (first) *yp = acc[im][in][j] + bias;
          else       *yp += acc[im][in][j];
        }
      }
    }
  }
}

// ------------------------------------------------------------- combine ----
__global__ __launch_bounds__(256) void k_combine(const float* __restrict__ Y,
    const int* __restrict__ row_of, const float* __restrict__ w_of,
    float* __restrict__ out)
{
  int t = blockIdx.x;
  int c = threadIdx.x * 4;
  int r0 = row_of[t * 2 + 0], r1 = row_of[t * 2 + 1];
  float w0 = w_of[t * 2 + 0], w1 = w_of[t * 2 + 1];
  F4 y0 = *(const F4*)(Y + (size_t)r0 * DOUT + c);
  F4 y1 = *(const F4*)(Y + (size_t)r1 * DOUT + c);
  F4 o;
  o.x = w0 * y0.x + w1 * y1.x;
  o.y = w0 * y0.y + w1 * y1.y;
  o.z = w0 * y0.z + w1 * y1.z;
  o.w = w0 * y0.w + w1 * y1.w;
  *(F4*)(out + (size_t)t * DOUT + c) = o;
}

// --------------------------------------------------------------- launch ----
extern "C" void kernel_launch(void* const* d_in, const int* in_sizes, int n_in,
                              void* d_out, int out_size, void* d_ws, size_t ws_size,
                              hipStream_t stream)
{
  const float* x  = (const float*)d_in[0];
  const float* W1 = (const float*)d_in[1];
  const float* b1 = (const float*)d_in[2];
  const float* W2 = (const float*)d_in[3];
  const float* b2 = (const float*)d_in[4];
  const float* Wg = (const float*)d_in[5];
  const float* bg = (const float*)d_in[6];
  float* out = (float*)d_out;
  (void)in_sizes; (void)n_in; (void)out_size; (void)ws_size;

  char* w = (char*)d_ws;
  size_t off = 0;
  auto take = [&](size_t bytes) -> void* {
    void* p = (void*)(w + off);
    off += (bytes + 255) & ~(size_t)255;
    return p;
  };
  int*   counts  = (int*)  take(NEXP * sizeof(int));
  int*   cursors = (int*)  take(NEXP * sizeof(int));
  int*   offsets = (int*)  take((NEXP + 1) * sizeof(int));
  int*   top_e   = (int*)  take((size_t)N_TOK * 2 * sizeof(int));
  float* top_w   = (float*)take((size_t)N_TOK * 2 * sizeof(float));
  int*   tok_idx = (int*)  take((size_t)NROUTE * sizeof(int));
  int*   row_of  = (int*)  take((size_t)N_TOK * 2 * sizeof(int));
  float* w_of    = (float*)take((size_t)N_TOK * 2 * sizeof(float));
  bf16*  xbf     = (bf16*) take((size_t)N_TOK * DIN * sizeof(bf16));
  bf16*  H       = (bf16*) take((size_t)NROUTE * HSLICE * sizeof(bf16));
  float* Y       = (float*)take((size_t)NROUTE * DOUT * sizeof(float));

  hipLaunchKernelGGL(k_zero,   dim3(1), dim3(32), 0, stream, counts, cursors);
  hipLaunchKernelGGL(k_gate,   dim3(N_TOK / 8), dim3(256), 0, stream,
                     x, Wg, bg, top_e, top_w, counts);
  hipLaunchKernelGGL(k_scan,   dim3(1), dim3(1), 0, stream, counts, offsets);
  hipLaunchKernelGGL(k_assign, dim3(N_TOK / 256), dim3(256), 0, stream,
                     top_e, top_w, offsets, cursors, tok_idx, row_of, w_of);
  hipLaunchKernelGGL(k_cvt,    dim3((N_TOK * DIN / 4) / 256), dim3(256), 0, stream,
                     x, xbf);
  for (int s = 0; s < NSLICES; ++s) {
    hipLaunchKernelGGL(k_ffn1, dim3(N_TOK / BM, HSLICE / BN, NEXP), dim3(256), 0, stream,
                       xbf, W1, b1, tok_idx, counts, offsets, H, s * HSLICE);
    hipLaunchKernelGGL(k_ffn2, dim3(N_TOK / BM, DOUT / BN, NEXP), dim3(256), 0, stream,
                       H, W2, b2, counts, offsets, Y, s * HSLICE, (s == 0) ? 1 : 0);
  }
  hipLaunchKernelGGL(k_combine, dim3(N_TOK), dim3(256), 0, stream,
                     Y, row_of, w_of, out);
}
